// KipfNet_res2_30210799960808
// MI455X (gfx1250) — matrix operand order repaired and checked by reading, not compiled, not run on hardware
//
#include <hip/hip_runtime.h>
#include <cstdint>
#include <cstddef>

// ---------------- problem constants ----------------
constexpr int N_ = 100000;
constexpr int E_ = 1600000;
constexpr int F_ = 64;     // input features
constexpr int C_ = 32;     // output classes
constexpr int KC = 8;      // Chebyshev order
constexpr int H_ = 36;     // hidden
constexpr int HP = 48;     // H padded to multiple of 16 (GEMM N-dim)
constexpr int KP = 64;     // K-dim padded to multiple of 32 (covers F=64 and H=36)
constexpr int K3 = 128;    // concat 3*H=108 padded to multiple of 32
constexpr float EPS = 1e-5f;

// ---------------- WMMA vector types ----------------
typedef __attribute__((ext_vector_type(16))) _Float16 v16h;
typedef __attribute__((ext_vector_type(8)))  _Float16 v8h;
typedef __attribute__((ext_vector_type(8)))  float    v8f;

// =====================================================================
// Graph preprocessing
// =====================================================================
__global__ void deg_kernel(const int* __restrict__ src, float* __restrict__ deg, int E) {
  int e = blockIdx.x * blockDim.x + threadIdx.x;
  if (e < E) atomicAdd(&deg[src[e]], 1.0f);
}

__global__ void dis_kernel(const float* __restrict__ deg, float* __restrict__ dis, int n) {
  int i = blockIdx.x * blockDim.x + threadIdx.x;
  if (i < n) {
    float d = deg[i];
    dis[i] = (d > 0.0f) ? rsqrtf(fmaxf(d, 1.0f)) : 0.0f;
  }
}

__global__ void edgew_kernel(const int* __restrict__ src, const int* __restrict__ dst,
                             const float* __restrict__ dis, float* __restrict__ ew, int E) {
  int e = blockIdx.x * blockDim.x + threadIdx.x;
  if (e < E) ew[e] = -dis[src[e]] * dis[dst[e]];
}

// =====================================================================
// SPMM: out[dst] += w * x[src]   (w==nullptr -> weight 1, for GIN agg)
// edge-parallel, 4 features per thread, float4 gather + f32 atomics (L2 resident)
// =====================================================================
__global__ void spmm_kernel(const float* __restrict__ x, const int* __restrict__ src,
                            const int* __restrict__ dst, const float* __restrict__ w,
                            float* __restrict__ out, int E, int F) {
  const int groups = F >> 2;
  long long gid = (long long)blockIdx.x * blockDim.x + threadIdx.x;
  if (gid >= (long long)E * groups) return;
  int e = (int)(gid / groups);
  int c = (int)(gid % groups) << 2;
  float wt = w ? w[e] : 1.0f;
  const float4 xv = *(const float4*)(x + (size_t)src[e] * F + c);
  float* o = out + (size_t)dst[e] * F + c;
  atomicAdd(o + 0, wt * xv.x);
  atomicAdd(o + 1, wt * xv.y);
  atomicAdd(o + 2, wt * xv.z);
  atomicAdd(o + 3, wt * xv.w);
}

// =====================================================================
// WMMA GEMM:  Cm[rows x Hpad] (+)= A[rows x Kpad](f16) * W (Wt = f16 [Hpad x Kpad], pre-transposed)
// One wave per 16x16 tile. Layouts per CDNA5 ISA 7.12.2:
//  A (16-bit 16x32): lane<16 -> M=lane, K={kk..kk+7, kk+16..kk+23}; lane>=16 -> K base +8
//  B (32x16): lane holds column n=lane&15, K same as A: {kk..kk+7, kk+16..kk+23}, +8 for lane>=16
//  C/D (f32 16x16): lane n=lane&15, VGPR r -> row (lane<16?0:8)+r
// =====================================================================
__global__ __launch_bounds__(32)
void gemm_wmma_kernel(const _Float16* __restrict__ A, const _Float16* __restrict__ Wt,
                      float* __restrict__ Cm, int Kpad, int Hpad, int accumulate) {
  const int lane = threadIdx.x & 31;
  const int hi   = lane >> 4;          // 0 | 1
  const int idx  = lane & 15;
  const int tm   = blockIdx.x * 16;
  const int n    = blockIdx.y * 16 + idx;
  const int m    = tm + idx;
  const int kbA  = hi * 8;
  const int kbB  = hi * 8;
  const int rb   = hi * 8;

  v8f c;
  if (accumulate) {
#pragma unroll
    for (int r = 0; r < 8; ++r) c[r] = Cm[(size_t)(tm + rb + r) * Hpad + n];
  } else {
#pragma unroll
    for (int r = 0; r < 8; ++r) c[r] = 0.0f;
  }

  const _Float16* arow = A  + (size_t)m * Kpad + kbA;
  const _Float16* brow = Wt + (size_t)n * Kpad + kbB;

  for (int kk = 0; kk < Kpad; kk += 32) {
    v8h a0 = *(const v8h*)(arow + kk);        // K = kb..kb+7
    v8h a1 = *(const v8h*)(arow + kk + 16);   // K = kb+16..kb+23
    v16h a;
#pragma unroll
    for (int j = 0; j < 8; ++j) { a[j] = a0[j]; a[j + 8] = a1[j]; }
    v8h b0 = *(const v8h*)(brow + kk);        // K = kbB..kbB+7
    v8h b1 = *(const v8h*)(brow + kk + 16);   // K = kbB+16..kbB+23
    v16h b;
#pragma unroll
    for (int j = 0; j < 8; ++j) { b[j] = b0[j]; b[j + 8] = b1[j]; }
    c = __builtin_amdgcn_wmma_f32_16x16x32_f16(false, a, false, b, (short)0, c,
                                               false, false);
  }

#pragma unroll
  for (int r = 0; r < 8; ++r) Cm[(size_t)(tm + rb + r) * Hpad + n] = c[r];
}

// =====================================================================
// Elementwise / conversion kernels
// =====================================================================
// f32 compact [rows x F] -> f16 padded [rows x kpad]
__global__ void cvt_pad_kernel(const float* __restrict__ in, _Float16* __restrict__ out,
                               int rows, int F, int kpad) {
  long long gid = (long long)blockIdx.x * blockDim.x + threadIdx.x;
  if (gid >= (long long)rows * F) return;
  int r = (int)(gid / F), c = (int)(gid % F);
  out[(size_t)r * kpad + c] = (_Float16)in[gid];
}

// Tx2 = 2*S - Tx0 ; write f32 compact and f16 padded
__global__ void cheb_combine_kernel(const float* __restrict__ s, const float* __restrict__ t0,
                                    float* __restrict__ t2f, _Float16* __restrict__ t2h,
                                    int rows, int F, int kpad) {
  long long gid = (long long)blockIdx.x * blockDim.x + threadIdx.x;
  if (gid >= (long long)rows * F) return;
  int r = (int)(gid / F), c = (int)(gid % F);
  float v = 2.0f * s[gid] - t0[gid];
  t2f[gid] = v;
  t2h[(size_t)r * kpad + c] = (_Float16)v;
}

// GIN: h = x + agg, to f16 flat (F==kpad==64)
__global__ void add_cvt_kernel(const float* __restrict__ x, const float* __restrict__ agg,
                               _Float16* __restrict__ out, long long n) {
  long long gid = (long long)blockIdx.x * blockDim.x + threadIdx.x;
  if (gid < n) out[gid] = (_Float16)(x[gid] + agg[gid]);
}

// relu(in + bias) -> f16 padded (GIN hidden)
__global__ void bias_relu_h_kernel(const float* __restrict__ in, const float* __restrict__ bias,
                                   _Float16* __restrict__ outH, int rows, int Hv, int sIn, int kpad) {
  long long gid = (long long)blockIdx.x * blockDim.x + threadIdx.x;
  if (gid >= (long long)rows * Hv) return;
  int r = (int)(gid / Hv), c = (int)(gid % Hv);
  float v = fmaxf(in[(size_t)r * sIn + c] + bias[c], 0.0f);
  outH[(size_t)r * kpad + c] = (_Float16)v;
}

// in-place relu(in + bias) in padded f32 buffer (GIN pre-BN)
__global__ void bias_relu_f_kernel(float* __restrict__ buf, const float* __restrict__ bias,
                                   int rows, int Hv, int sIn) {
  long long gid = (long long)blockIdx.x * blockDim.x + threadIdx.x;
  if (gid >= (long long)rows * Hv) return;
  int r = (int)(gid / Hv), c = (int)(gid % Hv);
  size_t o = (size_t)r * sIn + c;
  buf[o] = fmaxf(buf[o] + bias[c], 0.0f);
}

// per-column mean/biased-var (one block per column)
__global__ void colstats_kernel(const float* __restrict__ X, int rows, int stride,
                                float* __restrict__ mean, float* __restrict__ var) {
  __shared__ float ssum[256], ssq[256];
  const int c = blockIdx.x;
  float s = 0.0f, q = 0.0f;
  for (int r = threadIdx.x; r < rows; r += blockDim.x) {
    float v = X[(size_t)r * stride + c];
    s += v; q += v * v;
  }
  ssum[threadIdx.x] = s; ssq[threadIdx.x] = q;
  __syncthreads();
  for (int off = blockDim.x >> 1; off > 0; off >>= 1) {
    if ((int)threadIdx.x < off) {
      ssum[threadIdx.x] += ssum[threadIdx.x + off];
      ssq[threadIdx.x]  += ssq[threadIdx.x + off];
    }
    __syncthreads();
  }
  if (threadIdx.x == 0) {
    float m = ssum[0] / rows;
    mean[c] = m;
    var[c]  = ssq[0] / rows - m * m;
  }
}

// BN apply (+optional relu), fan out to: compact f32, padded f16, concat f16 slice
__global__ void bn_apply_kernel(const float* __restrict__ X, int rows, int Hv, int sIn,
                                const float* __restrict__ mean, const float* __restrict__ var,
                                const float* __restrict__ g, const float* __restrict__ be,
                                float* __restrict__ outF,
                                _Float16* __restrict__ outH, int kpadA,
                                _Float16* __restrict__ concatH, int coff,
                                int do_relu) {
  long long gid = (long long)blockIdx.x * blockDim.x + threadIdx.x;
  if (gid >= (long long)rows * Hv) return;
  int r = (int)(gid / Hv), c = (int)(gid % Hv);
  float v = X[(size_t)r * sIn + c];
  v = g[c] * (v - mean[c]) * rsqrtf(var[c] + EPS) + be[c];
  if (do_relu) v = fmaxf(v, 0.0f);
  if (outF)    outF[gid] = v;
  if (outH)    outH[(size_t)r * kpadA + c] = (_Float16)v;
  if (concatH) concatH[(size_t)r * K3 + coff + c] = (_Float16)v;
}

// W [Kin x Hout] f32 -> Wt [Hpad x Kpad] f16 (transposed, buffer pre-zeroed)
__global__ void wtcvt_kernel(const float* __restrict__ W, _Float16* __restrict__ Wt,
                             int Kin, int Hout, int Kpad) {
  int gid = blockIdx.x * blockDim.x + threadIdx.x;
  if (gid >= Kin * Hout) return;
  int k = gid / Hout, h = gid % Hout;
  Wt[(size_t)h * Kpad + k] = (_Float16)W[gid];
}

__global__ void bias_add_kernel(float* __restrict__ out, const float* __restrict__ b,
                                long long n, int Cc) {
  long long gid = (long long)blockIdx.x * blockDim.x + threadIdx.x;
  if (gid < n) out[gid] += b[(int)(gid % Cc)];
}

// =====================================================================
// Host orchestration
// =====================================================================
static inline int blk1(long long total) { return (int)((total + 255) / 256); }

extern "C" void kernel_launch(void* const* d_in, const int* in_sizes, int n_in,
                              void* d_out, int out_size, void* d_ws, size_t ws_size,
                              hipStream_t stream) {
  const float* x     = (const float*)d_in[0];
  const int*   ei    = (const int*)d_in[1];
  const float* W1_1  = (const float*)d_in[2];
  // b1_1 (d_in[3]) cancels exactly under training-mode BN
  const float* g1_1  = (const float*)d_in[4];
  const float* be1_1 = (const float*)d_in[5];
  const float* W1_2  = (const float*)d_in[6];
  // b1_2 (d_in[7]) cancels exactly under BN
  const float* g1_2  = (const float*)d_in[8];
  const float* be1_2 = (const float*)d_in[9];
  const float* gw1   = (const float*)d_in[10];
  const float* gb1   = (const float*)d_in[11];
  const float* gw2   = (const float*)d_in[12];
  const float* gb2   = (const float*)d_in[13];
  const float* g2    = (const float*)d_in[14];
  const float* be2   = (const float*)d_in[15];
  const float* W4    = (const float*)d_in[16];
  const float* b4    = (const float*)d_in[17];
  const int* src = ei;
  const int* dst = ei + E_;
  float* out = (float*)d_out;

  // -------- workspace carve (~220 MB) --------
  char* p = (char*)d_ws;
  auto carve = [&](size_t bytes) -> void* {
    void* r = (void*)p; p += (bytes + 255) & ~(size_t)255; return r;
  };
  float*    deg  = (float*)carve((size_t)N_ * 4);
  float*    dis  = (float*)carve((size_t)N_ * 4);
  float*    ew   = (float*)carve((size_t)E_ * 4);
  float*    B0   = (float*)carve((size_t)N_ * F_ * 4);
  float*    B1   = (float*)carve((size_t)N_ * F_ * 4);
  float*    B2   = (float*)carve((size_t)N_ * F_ * 4);
  float*    S    = (float*)carve((size_t)N_ * F_ * 4);
  _Float16* H0   = (_Float16*)carve((size_t)N_ * KP * 2);
  _Float16* H1   = (_Float16*)carve((size_t)N_ * KP * 2);
  _Float16* H2   = (_Float16*)carve((size_t)N_ * KP * 2);
  _Float16* XH   = (_Float16*)carve((size_t)N_ * KP * 2);
  float*    ACC  = (float*)carve((size_t)N_ * HP * 4);
  float*    X11C = (float*)carve((size_t)N_ * H_ * 4);
  _Float16* CCB  = (_Float16*)carve((size_t)N_ * K3 * 2);
  float*    MEAN = (float*)carve(HP * 4);
  float*    VAR  = (float*)carve(HP * 4);
  _Float16* Wt1  = (_Float16*)carve((size_t)KC * HP * KP * 2);
  _Float16* Wt2  = (_Float16*)carve((size_t)KC * HP * KP * 2);
  _Float16* GW1T = (_Float16*)carve((size_t)HP * KP * 2);
  _Float16* GW2T = (_Float16*)carve((size_t)HP * KP * 2);
  _Float16* W4T  = (_Float16*)carve((size_t)C_ * K3 * 2);

  // -------- weight prep: transpose+pad to f16 --------
  hipMemsetAsync(Wt1,  0, (size_t)KC * HP * KP * 2, stream);
  hipMemsetAsync(Wt2,  0, (size_t)KC * HP * KP * 2, stream);
  hipMemsetAsync(GW1T, 0, (size_t)HP * KP * 2, stream);
  hipMemsetAsync(GW2T, 0, (size_t)HP * KP * 2, stream);
  hipMemsetAsync(W4T,  0, (size_t)C_ * K3 * 2, stream);
  for (int k = 0; k < KC; ++k) {
    wtcvt_kernel<<<blk1(F_ * H_), 256, 0, stream>>>(W1_1 + (size_t)k * F_ * H_,
                                                    Wt1 + (size_t)k * HP * KP, F_, H_, KP);
    wtcvt_kernel<<<blk1(H_ * H_), 256, 0, stream>>>(W1_2 + (size_t)k * H_ * H_,
                                                    Wt2 + (size_t)k * HP * KP, H_, H_, KP);
  }
  wtcvt_kernel<<<blk1(F_ * H_), 256, 0, stream>>>(gw1, GW1T, F_, H_, KP);
  wtcvt_kernel<<<blk1(H_ * H_), 256, 0, stream>>>(gw2, GW2T, H_, H_, KP);
  wtcvt_kernel<<<blk1(3 * H_ * C_), 256, 0, stream>>>(W4, W4T, 3 * H_, C_, K3);

  // -------- graph prep --------
  hipMemsetAsync(deg, 0, (size_t)N_ * 4, stream);
  deg_kernel<<<blk1(E_), 256, 0, stream>>>(src, deg, E_);
  dis_kernel<<<blk1(N_), 256, 0, stream>>>(deg, dis, N_);
  edgew_kernel<<<blk1(E_), 256, 0, stream>>>(src, dst, dis, ew, E_);

  hipMemsetAsync(CCB, 0, (size_t)N_ * K3 * 2, stream);

  const dim3 gemmGrid(N_ / 16, HP / 16);
  const dim3 gemmGridC(N_ / 16, C_ / 16);

  // ================= ChebConv 1 (F=64 -> H, K=8) =================
  cvt_pad_kernel<<<blk1((long long)N_ * F_), 256, 0, stream>>>(x, XH, N_, F_, KP);
  gemm_wmma_kernel<<<gemmGrid, 32, 0, stream>>>(XH, Wt1 + 0, ACC, KP, HP, 0);

  hipMemsetAsync(B0, 0, (size_t)N_ * F_ * 4, stream);
  spmm_kernel<<<blk1((long long)E_ * (F_ / 4)), 256, 0, stream>>>(x, src, dst, ew, B0, E_, F_);
  cvt_pad_kernel<<<blk1((long long)N_ * F_), 256, 0, stream>>>(B0, H0, N_, F_, KP);
  gemm_wmma_kernel<<<gemmGrid, 32, 0, stream>>>(H0, Wt1 + (size_t)1 * HP * KP, ACC, KP, HP, 1);

  {
    const float* f2 = x;  const float* f1 = B0;
    float* fb[3] = {B0, B1, B2};
    _Float16* hb[3] = {H0, H1, H2};
    int slot = 1;
    for (int k = 2; k < KC; ++k) {
      float* tf = fb[slot]; _Float16* th = hb[slot];
      hipMemsetAsync(S, 0, (size_t)N_ * F_ * 4, stream);
      spmm_kernel<<<blk1((long long)E_ * (F_ / 4)), 256, 0, stream>>>(f1, src, dst, ew, S, E_, F_);
      cheb_combine_kernel<<<blk1((long long)N_ * F_), 256, 0, stream>>>(S, f2, tf, th, N_, F_, KP);
      gemm_wmma_kernel<<<gemmGrid, 32, 0, stream>>>(th, Wt1 + (size_t)k * HP * KP, ACC, KP, HP, 1);
      f2 = f1; f1 = tf; slot = (slot + 1) % 3;
    }
  }
  // BN + ReLU -> x1_1 (f32 compact for cheb2 spmm, f16 padded for cheb2 GEMM, concat[0:36))
  colstats_kernel<<<H_, 256, 0, stream>>>(ACC, N_, HP, MEAN, VAR);
  hipMemsetAsync(XH, 0, (size_t)N_ * KP * 2, stream);
  bn_apply_kernel<<<blk1((long long)N_ * H_), 256, 0, stream>>>(
      ACC, N_, H_, HP, MEAN, VAR, g1_1, be1_1, X11C, XH, KP, CCB, 0, 1);

  // ================= ChebConv 2 (H=36 -> H, K=8) =================
  hipMemsetAsync(H0, 0, (size_t)N_ * KP * 2, stream);
  hipMemsetAsync(H1, 0, (size_t)N_ * KP * 2, stream);
  hipMemsetAsync(H2, 0, (size_t)N_ * KP * 2, stream);

  gemm_wmma_kernel<<<gemmGrid, 32, 0, stream>>>(XH, Wt2 + 0, ACC, KP, HP, 0);

  hipMemsetAsync(B0, 0, (size_t)N_ * H_ * 4, stream);
  spmm_kernel<<<blk1((long long)E_ * (H_ / 4)), 256, 0, stream>>>(X11C, src, dst, ew, B0, E_, H_);
  cvt_pad_kernel<<<blk1((long long)N_ * H_), 256, 0, stream>>>(B0, H0, N_, H_, KP);
  gemm_wmma_kernel<<<gemmGrid, 32, 0, stream>>>(H0, Wt2 + (size_t)1 * HP * KP, ACC, KP, HP, 1);

  {
    const float* f2 = X11C;  const float* f1 = B0;
    float* fb[3] = {B0, B1, B2};
    _Float16* hb[3] = {H0, H1, H2};
    int slot = 1;
    for (int k = 2; k < KC; ++k) {
      float* tf = fb[slot]; _Float16* th = hb[slot];
      hipMemsetAsync(S, 0, (size_t)N_ * H_ * 4, stream);
      spmm_kernel<<<blk1((long long)E_ * (H_ / 4)), 256, 0, stream>>>(f1, src, dst, ew, S, E_, H_);
      cheb_combine_kernel<<<blk1((long long)N_ * H_), 256, 0, stream>>>(S, f2, tf, th, N_, H_, KP);
      gemm_wmma_kernel<<<gemmGrid, 32, 0, stream>>>(th, Wt2 + (size_t)k * HP * KP, ACC, KP, HP, 1);
      f2 = f1; f1 = tf; slot = (slot + 1) % 3;
    }
  }
  colstats_kernel<<<H_, 256, 0, stream>>>(ACC, N_, HP, MEAN, VAR);
  bn_apply_kernel<<<blk1((long long)N_ * H_), 256, 0, stream>>>(
      ACC, N_, H_, HP, MEAN, VAR, g1_2, be1_2, nullptr, nullptr, KP, CCB, H_, 1);

  // ================= GIN branch =================
  hipMemsetAsync(S, 0, (size_t)N_ * F_ * 4, stream);
  spmm_kernel<<<blk1((long long)E_ * (F_ / 4)), 256, 0, stream>>>(x, src, dst, nullptr, S, E_, F_);
  add_cvt_kernel<<<blk1((long long)N_ * F_), 256, 0, stream>>>(x, S, H0, (long long)N_ * F_);
  gemm_wmma_kernel<<<gemmGrid, 32, 0, stream>>>(H0, GW1T, ACC, KP, HP, 0);
  hipMemsetAsync(H1, 0, (size_t)N_ * KP * 2, stream);
  bias_relu_h_kernel<<<blk1((long long)N_ * H_), 256, 0, stream>>>(ACC, gb1, H1, N_, H_, HP, KP);
  gemm_wmma_kernel<<<gemmGrid, 32, 0, stream>>>(H1, GW2T, ACC, KP, HP, 0);
  bias_relu_f_kernel<<<blk1((long long)N_ * H_), 256, 0, stream>>>(ACC, gb2, N_, H_, HP);
  colstats_kernel<<<H_, 256, 0, stream>>>(ACC, N_, HP, MEAN, VAR);
  bn_apply_kernel<<<blk1((long long)N_ * H_), 256, 0, stream>>>(
      ACC, N_, H_, HP, MEAN, VAR, g2, be2, nullptr, nullptr, KP, CCB, 2 * H_, 0);

  // ================= final linear: concat(108->128) @ W4 + b4 =================
  gemm_wmma_kernel<<<gemmGridC, 32, 0, stream>>>(CCB, W4T, out, K3, C_, 0);
  bias_add_kernel<<<blk1((long long)N_ * C_), 256, 0, stream>>>(out, b4, (long long)N_ * C_, C_);
}
